// EncoderBlock_74655121539548
// MI455X (gfx1250) — compile-verified, matrix operand order checked by reading, not run on hardware
//
#include <hip/hip_runtime.h>
#include <hip/hip_bf16.h>
#include <math.h>

// ---------------------------------------------------------------------------
// Types
// ---------------------------------------------------------------------------
typedef __bf16 bf16;
typedef __attribute__((ext_vector_type(16))) __bf16 bf16x16;
typedef __attribute__((ext_vector_type(8)))  float  f32x8;
typedef __attribute__((ext_vector_type(4)))  int    v4i;

struct alignas(16) U128 { unsigned int x[4]; };
union FragBF { U128 u[2]; bf16x16 v; };
union Pack8  { U128 u; bf16 h[8]; };

#define EPS_LN 1e-6f
#define EPS_IN 1e-5f

// CDNA5 async global->LDS path (ASYNCcnt tracked), guarded so absence of the
// builtin degrades to plain b128 staging.
#if defined(__gfx1250__) && __has_builtin(__builtin_amdgcn_global_load_async_to_lds_b128)
#define HAVE_ASYNC_LDS 1
#endif
#define ASG __attribute__((address_space(1)))
#define AS3 __attribute__((address_space(3)))

#ifdef HAVE_ASYNC_LDS
static __device__ __forceinline__ void async_copy_b128(const bf16* gp, bf16* lp) {
  // GLOBAL_LOAD_ASYNC_TO_LDS_B128: per-lane 16B global -> LDS, ASYNCcnt.
  __builtin_amdgcn_global_load_async_to_lds_b128(
      (ASG v4i*)gp, (AS3 v4i*)lp, 0, 0);
}
#endif

// ---------------------------------------------------------------------------
// Block reductions (blockDim.x == 256, 8 wave32 waves)
// ---------------------------------------------------------------------------
static __device__ __forceinline__ float blk_sum(float v, float* sbuf) {
#pragma unroll
  for (int o = 16; o > 0; o >>= 1) v += __shfl_xor(v, o, 32);
  if ((threadIdx.x & 31) == 0) sbuf[threadIdx.x >> 5] = v;
  __syncthreads();
  if (threadIdx.x == 0) {
    float t = 0.f;
#pragma unroll
    for (int i = 0; i < 8; ++i) t += sbuf[i];
    sbuf[0] = t;
  }
  __syncthreads();
  float r = sbuf[0];
  __syncthreads();
  return r;
}

static __device__ __forceinline__ float blk_max(float v, float* sbuf) {
#pragma unroll
  for (int o = 16; o > 0; o >>= 1) v = fmaxf(v, __shfl_xor(v, o, 32));
  if ((threadIdx.x & 31) == 0) sbuf[threadIdx.x >> 5] = v;
  __syncthreads();
  if (threadIdx.x == 0) {
    float t = sbuf[0];
#pragma unroll
    for (int i = 1; i < 8; ++i) t = fmaxf(t, sbuf[i]);
    sbuf[0] = t;
  }
  __syncthreads();
  float r = sbuf[0];
  __syncthreads();
  return r;
}

// ---------------------------------------------------------------------------
// WMMA GEMM:  D = epi( A(MxK) * B(KxN) )
//   TA: A stored K-major as (K x M) row-major (physically transposed)
//   TB: B stored N-major as (N x K) row-major (physically transposed)
// Block: 256 threads (8 waves). Tile 128(M) x 128(N) x 32(K).
// Wave grid 4x2; each wave computes a 32x64 patch = 2x4 16x16x32 WMMA tiles.
// ---------------------------------------------------------------------------
enum { EPI_F32 = 0, EPI_BF16 = 1, EPI_SCALE_F32 = 2, EPI_RES_F32 = 3,
       EPI_BIAS_GELU_BF16 = 4, EPI_BIAS_RES_F32 = 5 };

template <bool TA, bool TB, int EPI>
__global__ __launch_bounds__(256) void wmma_gemm(
    const bf16* __restrict__ A, const bf16* __restrict__ Bm,
    void* __restrict__ Dv,
    const float* __restrict__ bias, const float* __restrict__ res,
    int M, int N, int K, int lda, int ldb, int ldd,
    long sA, long sB, long sD, long sRes, float scale) {
  constexpr int BM = 128, BN = 128, BK = 32, LSH = 48;  // LSH: padded LDS row
  __shared__ bf16 As[BM * LSH];
  __shared__ bf16 Bs[BN * LSH];

  const int tid  = threadIdx.x;
  const int lane = tid & 31;
  const int wid  = tid >> 5;
  const int wm   = wid & 3;        // wave row (0..3) -> 32 rows each
  const int wn   = wid >> 2;       // wave col (0..1) -> 64 cols each
  const int z    = blockIdx.z;
  const int m0   = blockIdx.y * BM;
  const int n0   = blockIdx.x * BN;

  A  += (long)z * sA;
  Bm += (long)z * sB;

  f32x8 acc[2][4] = {};
  const int nkt = K / BK;  // all contraction dims are multiples of 32

  for (int kt = 0; kt < nkt; ++kt) {
    const int k0 = kt * BK;
    __syncthreads();
    // ---- stage A tile into LDS: As[m * LSH + k] --------------------------
    if (!TA) {
#pragma unroll
      for (int g = 0; g < 2; ++g) {
        int gid = tid + g * 256;      // 0..511
        int m   = gid >> 2;           // 0..127
        int k   = (gid & 3) * 8;      // 0,8,16,24
        int gm  = m0 + m;
        if (gm < M) {
          const bf16* gp = A + (long)gm * lda + (k0 + k);
#ifdef HAVE_ASYNC_LDS
          async_copy_b128(gp, &As[m * LSH + k]);
#else
          *(U128*)(&As[m * LSH + k]) = *(const U128*)gp;
#endif
          if (kt + 1 < nkt) __builtin_prefetch(gp + BK, 0, 1);
        } else {
          *(U128*)(&As[m * LSH + k]) = U128{};
        }
      }
    } else {  // A stored (K x M): gather transposed
#pragma unroll
      for (int g = 0; g < 2; ++g) {
        int gid = tid + g * 256;
        int m   = gid >> 2;
        int kq  = (gid & 3) * 8;
        int gm  = m0 + m;
#pragma unroll
        for (int e = 0; e < 8; ++e) {
          bf16 val = (bf16)0.0f;
          if (gm < M) val = A[(long)(k0 + kq + e) * lda + gm];
          As[m * LSH + kq + e] = val;
        }
      }
    }
    // ---- stage B tile into LDS: Bs[n * LSH + k] --------------------------
    if (!TB) {  // B stored (K x N): read 8-col rows, scatter into n-major LDS
#pragma unroll
      for (int g = 0; g < 2; ++g) {
        int gid = tid + g * 256;     // 0..511
        int k   = gid >> 4;          // 0..31
        int n   = (gid & 15) * 8;    // 0..120
        Pack8 p; p.u = U128{};
        if (n0 + n < N)
          p.u = *(const U128*)(Bm + (long)(k0 + k) * ldb + (n0 + n));
#pragma unroll
        for (int e = 0; e < 8; ++e) Bs[(n + e) * LSH + k] = p.h[e];
      }
    } else {    // B stored (N x K): contiguous along k -> async b128 copy
#pragma unroll
      for (int g = 0; g < 2; ++g) {
        int gid = tid + g * 256;
        int n   = gid >> 2;          // 0..127
        int k   = (gid & 3) * 8;
        if (n0 + n < N) {
          const bf16* gp = Bm + (long)(n0 + n) * ldb + (k0 + k);
#ifdef HAVE_ASYNC_LDS
          async_copy_b128(gp, &Bs[n * LSH + k]);
#else
          *(U128*)(&Bs[n * LSH + k]) = *(const U128*)gp;
#endif
          if (kt + 1 < nkt) __builtin_prefetch(gp + BK, 0, 1);
        } else {
          *(U128*)(&Bs[n * LSH + k]) = U128{};
        }
      }
    }
#ifdef HAVE_ASYNC_LDS
#if __has_builtin(__builtin_amdgcn_s_wait_asynccnt)
    __builtin_amdgcn_s_wait_asynccnt(0);
#else
    asm volatile("s_wait_asynccnt 0x0" ::: "memory");
#endif
#endif
    __syncthreads();
    // ---- WMMA compute ----------------------------------------------------
    // 16-bit A-matrix 16x32 layout: lanes 0-15 hold K 0-7 / 16-23,
    // lanes 16-31 hold K 8-15 / 24-31 (two 16B halves per lane).
    const int kb = (lane < 16) ? 0 : 8;
    const int lr = lane & 15;
    FragBF fa[2], fb[4];
#pragma unroll
    for (int i = 0; i < 2; ++i) {
      int row = wm * 32 + i * 16 + lr;
      fa[i].u[0] = *(const U128*)(&As[row * LSH + kb]);
      fa[i].u[1] = *(const U128*)(&As[row * LSH + kb + 16]);
    }
#pragma unroll
    for (int j = 0; j < 4; ++j) {
      int col = wn * 64 + j * 16 + lr;
      fb[j].u[0] = *(const U128*)(&Bs[col * LSH + kb]);
      fb[j].u[1] = *(const U128*)(&Bs[col * LSH + kb + 16]);
    }
#pragma unroll
    for (int i = 0; i < 2; ++i)
#pragma unroll
      for (int j = 0; j < 4; ++j)
        acc[i][j] = __builtin_amdgcn_wmma_f32_16x16x32_bf16(
            false, fa[i].v, false, fb[j].v, (short)0, acc[i][j], false, false);
  }

  // ---- epilogue: C/D layout lanes 0-15 -> M = vgpr, lanes 16-31 -> M = vgpr+8
  float* Df = (float*)Dv + (long)z * sD;
  bf16*  Db = (bf16*)Dv  + (long)z * sD;
  const float* R = res ? (res + (long)z * sRes) : nullptr;
  const int rh  = (lane >> 4) * 8;
  const int col = lane & 15;
#pragma unroll
  for (int i = 0; i < 2; ++i)
#pragma unroll
    for (int j = 0; j < 4; ++j) {
      int bm = m0 + wm * 32 + i * 16;
      int bn = n0 + wn * 64 + j * 16;
#pragma unroll
      for (int v = 0; v < 8; ++v) {
        int gm = bm + rh + v, gn = bn + col;
        if (gm < M && gn < N) {
          float x = acc[i][j][v];
          long  o = (long)gm * ldd + gn;
          if constexpr (EPI == EPI_F32) {
            Df[o] = x;
          } else if constexpr (EPI == EPI_BF16) {
            Db[o] = (bf16)x;
          } else if constexpr (EPI == EPI_SCALE_F32) {
            Df[o] = x * scale;
          } else if constexpr (EPI == EPI_RES_F32) {
            Df[o] = x + R[o];
          } else if constexpr (EPI == EPI_BIAS_GELU_BF16) {
            float t = x + bias[gn];
            t = 0.5f * t * (1.0f + erff(t * 0.70710678118654752f));
            Db[o] = (bf16)t;
          } else {  // EPI_BIAS_RES_F32
            Df[o] = x + bias[gn] + R[o];
          }
        }
      }
    }
}

// ---------------------------------------------------------------------------
// Fused per-segment LayerNorm (4 branches) + LayerNorm of the concat row.
// One block per token (8192 blocks); row length 960 staged in LDS.
// ---------------------------------------------------------------------------
__global__ __launch_bounds__(256) void ln4_kernel(
    const float* __restrict__ e0, const float* __restrict__ e1,
    const float* __restrict__ e2, const float* __restrict__ e3,
    const float* __restrict__ g0, const float* __restrict__ b0,
    const float* __restrict__ g1, const float* __restrict__ b1,
    const float* __restrict__ g2, const float* __restrict__ b2,
    const float* __restrict__ g3, const float* __restrict__ b3,
    const float* __restrict__ gall, const float* __restrict__ ball,
    bf16* __restrict__ x0, bf16* __restrict__ x1,
    bf16* __restrict__ x2, bf16* __restrict__ x3,
    bf16* __restrict__ xall) {
  __shared__ float row[960];
  __shared__ float sbuf[8];
  __shared__ float segstat[8];
  const int t   = blockIdx.x;
  const int tid = threadIdx.x;
  const float* e[4]  = {e0, e1, e2, e3};
  const float* g[4]  = {g0, g1, g2, g3};
  const float* bb[4] = {b0, b1, b2, b3};
  bf16* xs[4] = {x0, x1, x2, x3};
  const int dims[4] = {64, 128, 256, 512};
  const int offs[4] = {0, 64, 192, 448};
#pragma unroll
  for (int s = 0; s < 4; ++s)
    for (int c = tid; c < dims[s]; c += 256)
      row[offs[s] + c] = e[s][(long)t * dims[s] + c];
  __syncthreads();
  for (int s = 0; s < 4; ++s) {
    float p = 0.f, q = 0.f;
    for (int c = tid; c < dims[s]; c += 256) {
      float v = row[offs[s] + c]; p += v; q += v * v;
    }
    p = blk_sum(p, sbuf);
    q = blk_sum(q, sbuf);
    if (tid == 0) { segstat[2 * s] = p; segstat[2 * s + 1] = q; }
  }
  __syncthreads();
  float sm[4], sr[4], sa = 0.f, qa = 0.f;
#pragma unroll
  for (int s = 0; s < 4; ++s) {
    sa += segstat[2 * s]; qa += segstat[2 * s + 1];
    float m   = segstat[2 * s] / (float)dims[s];
    float var = segstat[2 * s + 1] / (float)dims[s] - m * m;
    sm[s] = m; sr[s] = rsqrtf(var + EPS_LN);
  }
  float ma = sa / 960.f;
  float ra = rsqrtf(qa / 960.f - ma * ma + EPS_LN);
  for (int c = tid; c < 960; c += 256) {
    int s  = (c < 64) ? 0 : (c < 192) ? 1 : (c < 448) ? 2 : 3;
    int lc = c - offs[s];
    float v = row[c];
    xs[s][(long)t * dims[s] + lc] =
        (bf16)((v - sm[s]) * sr[s] * g[s][lc] + bb[s][lc]);
    xall[(long)t * 960 + c] = (bf16)((v - ma) * ra * gall[c] + ball[c]);
  }
}

// InstanceNorm stats per (h,b): mean and 1/sqrt(var+eps) over d*KV.
__global__ __launch_bounds__(256) void inorm_stats_kernel(
    const float* __restrict__ S, float* __restrict__ stats, int len) {
  __shared__ float sbuf[8];
  const long base = (long)blockIdx.x * len;
  float p = 0.f, q = 0.f;
  for (int i = threadIdx.x; i < len; i += 256) {
    float v = S[base + i]; p += v; q += v * v;
  }
  p = blk_sum(p, sbuf);
  q = blk_sum(q, sbuf);
  if (threadIdx.x == 0) {
    float m = p / (float)len;
    stats[2 * blockIdx.x]     = m;
    stats[2 * blockIdx.x + 1] = rsqrtf(q / (float)len - m * m + EPS_IN);
  }
}

// Normalize with instance stats, softmax over KV=960, emit bf16 A matrix.
__global__ __launch_bounds__(256) void inorm_softmax_kernel(
    const float* __restrict__ S, const float* __restrict__ stats,
    bf16* __restrict__ Aout, int d) {
  __shared__ float rowv[960];
  __shared__ float sbuf[8];
  const int  row  = blockIdx.x;
  const int  z    = row / d;
  const long base = (long)row * 960;
  const float m = stats[2 * z], r = stats[2 * z + 1];
  float mx = -3.4e38f;
  for (int c = threadIdx.x; c < 960; c += 256) {
    float tv = (S[base + c] - m) * r;
    rowv[c] = tv;
    mx = fmaxf(mx, tv);
  }
  mx = blk_max(mx, sbuf);
  float sum = 0.f;
  for (int c = threadIdx.x; c < 960; c += 256) {
    float ev = expf(rowv[c] - mx);
    rowv[c] = ev; sum += ev;
  }
  sum = blk_sum(sum, sbuf);
  float inv = 1.f / sum;
  for (int c = threadIdx.x; c < 960; c += 256)
    Aout[base + c] = (bf16)(rowv[c] * inv);
}

// Cm[b*N+n][m] = bf16( mean_h C[((h*B+b)*d+m)*N + n] )   (B=8, N=1024, H=4)
__global__ __launch_bounds__(256) void mean_heads_kernel(
    const float* __restrict__ C, bf16* __restrict__ Cm, int d) {
  const long total = (long)8192 * d;
  for (long idx = (long)blockIdx.x * 256 + threadIdx.x; idx < total;
       idx += (long)gridDim.x * 256) {
    int tok = (int)(idx / d), m = (int)(idx % d);
    int b = tok >> 10, n = tok & 1023;
    float s = 0.f;
#pragma unroll
    for (int h = 0; h < 4; ++h)
      s += C[(((long)(h * 8 + b) * d + m) << 10) + n];
    Cm[idx] = (bf16)(0.25f * s);
  }
}

// LayerNorm of attn_out rows (f32 in d_out) -> bf16 FFN input.
__global__ __launch_bounds__(256) void ffn_ln_kernel(
    const float* __restrict__ src, const float* __restrict__ g,
    const float* __restrict__ b, bf16* __restrict__ y, int d) {
  __shared__ float sbuf[8];
  const long base = (long)blockIdx.x * d;
  float p = 0.f, q = 0.f;
  for (int c = threadIdx.x; c < d; c += 256) {
    float v = src[base + c]; p += v; q += v * v;
  }
  p = blk_sum(p, sbuf);
  q = blk_sum(q, sbuf);
  float m = p / (float)d;
  float r = rsqrtf(q / (float)d - m * m + EPS_LN);
  for (int c = threadIdx.x; c < d; c += 256)
    y[base + c] = (bf16)((src[base + c] - m) * r * g[c] + b[c]);
}

// f32 -> bf16 weight conversion.
__global__ void cvt_bf16_kernel(const float* __restrict__ s,
                                bf16* __restrict__ d, long n) {
  for (long i = (long)blockIdx.x * blockDim.x + threadIdx.x; i < n;
       i += (long)gridDim.x * blockDim.x)
    d[i] = (bf16)s[i];
}

// ---------------------------------------------------------------------------
// Orchestration
// ---------------------------------------------------------------------------
extern "C" void kernel_launch(void* const* d_in, const int* in_sizes, int n_in,
                              void* d_out, int out_size, void* d_ws,
                              size_t ws_size, hipStream_t stream) {
  (void)in_sizes; (void)n_in; (void)out_size; (void)ws_size;
  const int Hh = 4, KVn = 960, Nn = 1024, T = 8192;
  const int dims[4] = {64, 128, 256, 512};
  const long outoff[4] = {0, (long)T * 64, (long)T * 192, (long)T * 448};

  const float* emb[4];
  for (int i = 0; i < 4; ++i) emb[i] = (const float*)d_in[i];
  auto pp = [&](int idx) { return (const float*)d_in[idx]; };
  // param indices: branch base = 4 + 10*i:
  //  +0 ln_g +1 ln_b +2 WQ +3 WO +4 ffn_g +5 ffn_b +6 W1 +7 b1 +8 W2 +9 b2
  //  44 gall, 45 ball, 46 WK, 47 WV

  char* w = (char*)d_ws; size_t off = 0;
  auto alloc = [&](size_t bytes) -> void* {
    off = (off + 255) & ~(size_t)255;
    void* p = w + off; off += bytes; return p;
  };
  bf16* xall = (bf16*)alloc((size_t)T * KVn * 2);
  bf16* xs[4];
  for (int i = 0; i < 4; ++i) xs[i] = (bf16*)alloc((size_t)T * dims[i] * 2);
  bf16*  Kh    = (bf16*)alloc((size_t)Hh * T * KVn * 2);   // [H][B*N][KV]
  bf16*  Vh    = (bf16*)alloc((size_t)Hh * T * KVn * 2);
  bf16*  Qh    = (bf16*)alloc((size_t)Hh * T * 512 * 2);   // reused per branch
  float* Sb    = (float*)alloc((size_t)32 * 512 * KVn * 4);
  float* stats = (float*)alloc(1024);
  bf16*  Asoft = (bf16*)alloc((size_t)32 * 512 * KVn * 2);
  float* Cb    = (float*)alloc((size_t)32 * 512 * 1024 * 4);
  bf16*  Cmb   = (bf16*)alloc((size_t)T * 512 * 2);
  bf16*  Yb    = (bf16*)alloc((size_t)T * 512 * 2);
  bf16*  H1    = (bf16*)alloc((size_t)T * 2048 * 2);
  bf16*  WKb   = (bf16*)alloc((size_t)Hh * KVn * KVn * 2);
  bf16*  WVb   = (bf16*)alloc((size_t)Hh * KVn * KVn * 2);
  bf16*  WQb   = (bf16*)alloc((size_t)Hh * 512 * 512 * 2); // reused per branch
  bf16*  WOb   = (bf16*)alloc((size_t)512 * 512 * 2);
  bf16*  W1b   = (bf16*)alloc((size_t)512 * 2048 * 2);
  bf16*  W2b   = (bf16*)alloc((size_t)2048 * 512 * 2);

  const dim3 blk(256);
  const float scaleS = 1.0f / sqrtf(960.f);

  // --- LayerNorms (per-segment + concat) ---------------------------------
  ln4_kernel<<<T, blk, 0, stream>>>(
      emb[0], emb[1], emb[2], emb[3],
      pp(4), pp(5), pp(14), pp(15), pp(24), pp(25), pp(34), pp(35),
      pp(44), pp(45), xs[0], xs[1], xs[2], xs[3], xall);

  // --- weights to bf16 ----------------------------------------------------
  cvt_bf16_kernel<<<4096, blk, 0, stream>>>(pp(46), WKb, (long)Hh * KVn * KVn);
  cvt_bf16_kernel<<<4096, blk, 0, stream>>>(pp(47), WVb, (long)Hh * KVn * KVn);

  // --- K/V projections: Kh/Vh[h][tok][kv] = LN(all) @ WK/WV[h] ------------
  {
    dim3 g((KVn + 127) / 128, T / 128, Hh);
    wmma_gemm<false, false, EPI_BF16><<<g, blk, 0, stream>>>(
        xall, WKb, Kh, nullptr, nullptr, T, KVn, KVn, KVn, KVn, KVn,
        0L, (long)KVn * KVn, (long)T * KVn, 0L, 1.f);
    wmma_gemm<false, false, EPI_BF16><<<g, blk, 0, stream>>>(
        xall, WVb, Vh, nullptr, nullptr, T, KVn, KVn, KVn, KVn, KVn,
        0L, (long)KVn * KVn, (long)T * KVn, 0L, 1.f);
  }

  // --- per-branch attention + FFN ----------------------------------------
  for (int i = 0; i < 4; ++i) {
    const int d = dims[i], dm = 4 * d, base = 4 + 10 * i;
    float* outI = (float*)d_out + outoff[i];

    cvt_bf16_kernel<<<1024, blk, 0, stream>>>(pp(base + 2), WQb, (long)Hh * d * d);
    cvt_bf16_kernel<<<1024, blk, 0, stream>>>(pp(base + 3), WOb, (long)d * d);
    cvt_bf16_kernel<<<1024, blk, 0, stream>>>(pp(base + 6), W1b, (long)d * dm);
    cvt_bf16_kernel<<<1024, blk, 0, stream>>>(pp(base + 8), W2b, (long)dm * d);

    // Qh[h][tok][d] = x_i @ WQ[h]
    {
      dim3 g((d + 127) / 128, T / 128, Hh);
      wmma_gemm<false, false, EPI_BF16><<<g, blk, 0, stream>>>(
          xs[i], WQb, Qh, nullptr, nullptr, T, d, d, d, d, d,
          0L, (long)d * d, (long)T * d, 0L, 1.f);
    }
    // S[z][d][KV] = scale * Qh^T(n,d) . Kh(n,KV), contraction over n=1024
    {
      dim3 g((KVn + 127) / 128, (d + 127) / 128, 32);
      wmma_gemm<true, false, EPI_SCALE_F32><<<g, blk, 0, stream>>>(
          Qh, Kh, Sb, nullptr, nullptr, d, KVn, Nn, d, KVn, KVn,
          (long)Nn * d, (long)Nn * KVn, (long)d * KVn, 0L, scaleS);
    }
    inorm_stats_kernel<<<32, blk, 0, stream>>>(Sb, stats, d * KVn);
    inorm_softmax_kernel<<<32 * d, blk, 0, stream>>>(Sb, stats, Asoft, d);

    // C[z][d][N] = A(d,KV) . Vh^T(KV,N)  (Vh stored [N][KV] -> TB)
    {
      dim3 g((Nn + 127) / 128, (d + 127) / 128, 32);
      wmma_gemm<false, true, EPI_F32><<<g, blk, 0, stream>>>(
          Asoft, Vh, Cb, nullptr, nullptr, d, Nn, KVn, KVn, KVn, Nn,
          (long)d * KVn, (long)Nn * KVn, (long)d * Nn, 0L, 1.f);
    }
    mean_heads_kernel<<<4096, blk, 0, stream>>>(Cb, Cmb, d);

    // attn_out = Cm @ WO + emb_i  (f32 into d_out)
    {
      dim3 g((d + 127) / 128, T / 128, 1);
      wmma_gemm<false, false, EPI_RES_F32><<<g, blk, 0, stream>>>(
          Cmb, WOb, outI, nullptr, emb[i], T, d, d, d, d, d,
          0L, 0L, 0L, 0L, 1.f);
    }
    // FFN
    ffn_ln_kernel<<<T, blk, 0, stream>>>(outI, pp(base + 4), pp(base + 5), Yb, d);
    {
      dim3 g((dm + 127) / 128, T / 128, 1);
      wmma_gemm<false, false, EPI_BIAS_GELU_BF16><<<g, blk, 0, stream>>>(
          Yb, W1b, H1, pp(base + 7), nullptr, T, dm, d, d, dm, dm,
          0L, 0L, 0L, 0L, 1.f);
    }
    {
      dim3 g((d + 127) / 128, T / 128, 1);
      wmma_gemm<false, false, EPI_BIAS_RES_F32><<<g, blk, 0, stream>>>(
          H1, W2b, outI, pp(base + 9), outI, T, d, dm, dm, d, d,
          0L, 0L, 0L, 0L, 1.f);
    }
  }
}
